// STBNB_90177133347599
// MI455X (gfx1250) — matrix operand order repaired and checked by reading, not compiled, not run on hardware
//
#include <hip/hip_runtime.h>
#include <hip/hip_bf16.h>
#include <math.h>

#define N_    20000
#define EMB_  128
#define F_    64
#define SL_   24
#define E_    30000
#define RAW_  384
#define H_    64
#define NT_   2

// ---------------- workspace layout (float offsets) ----------------
// raw region is dead after the first GEMM; AT/A/B buffers alias into it.
static const size_t RAW_OFF    = 0;            // [N,SL,384]  184,320,000
static const size_t AT_OFF     = 0;            // [N,SL,128]   61,440,000
static const size_t A_OFF      = 61440000;     // [N,SL,128]
static const size_t B_OFF      = 122880000;    // [N,SL,128]
static const size_t X_OFF      = 184320000;    // [N,SL,128]
static const size_t DEN_OFF    = 245760000;    // den_s[SL,N] then den_e[SL,N] = 960,000
static const size_t APPEAR_OFF = 246720000;    // [N]
static const size_t DM_OFF     = 246740000;    // [N,128]
static const size_t GI_OFF     = 249300000;    // [N,384]
static const size_t GH_OFF     = 256980000;    // [N,384]
static const size_t Q_OFF      = 264660000;    // [N,128]
static const size_t KB_OFF     = 267220000;    // [SL,N,128]  61,440,000
static const size_t STAT_OFF   = 328660000;    // 16: per task {sum,sumsq,mu,sd}
static const size_t WV1_OFF    = 328660016;    // wv1[64], bv1[64]
static const size_t WB_OFF     = 328660144;    // bf16 weight area (as __bf16)

typedef __attribute__((ext_vector_type(16))) __bf16 v16bf;
typedef __attribute__((ext_vector_type(8)))  float  v8f;

__device__ __forceinline__ unsigned bfbits(float f) {
  unsigned u = __builtin_bit_cast(unsigned, f);
  return ((u + 0x7FFFu + ((u >> 16) & 1u)) >> 16) & 0xFFFFu;
}

__device__ __forceinline__ __bf16 f2bf(float f) {
  return __builtin_bit_cast(__bf16, (unsigned short)bfbits(f));
}

__device__ __forceinline__ unsigned pack2(float lo, float hi) {
  return (bfbits(hi) << 16) | bfbits(lo);
}

__device__ __forceinline__ float wave_sum(float v) {
  #pragma unroll
  for (int off = 16; off > 0; off >>= 1) v += __shfl_xor(v, off, 32);
  return v;
}

__device__ __forceinline__ float gelu_exact(float x) {
  return 0.5f * x * (1.0f + erff(x * 0.70710678118654752f));
}

// ---------------- weight repack: f32 [K,Nc] -> bf16 in WMMA-B lane layout ----------------
// dst layout: [ct = col/16][kb = k/32][lane 0..31][e 0..15]
// lane: col-in-tile = lane&15, khalf = lane>>4; element e -> kr = kb*32 + 16*khalf + e
__global__ void __launch_bounds__(256)
repack_w(const float* __restrict__ W, __bf16* __restrict__ dst, int K, int Nc) {
  int i = blockIdx.x * 256 + threadIdx.x;
  if (i >= K * Nc) return;
  int e    = i & 15;
  int lane = (i >> 4) & 31;
  int rem  = i >> 9;                 // ct*(K/32) + kb
  int nkb  = K >> 5;
  int kb   = rem % nkb;
  int ct   = rem / nkb;
  int col  = ct * 16 + (lane & 15);
  int kr   = kb * 32 + 16 * (lane >> 4) + e;
  dst[i] = f2bf(W[(size_t)kr * Nc + col]);
}

// ---------------- edge aggregation (segment sums via L2 float atomics) ----------------
__global__ void __launch_bounds__(256)
agg_kernel(const int* __restrict__ st, const int* __restrict__ en,
           const float* __restrict__ ts, const float* __restrict__ bt,
           const float* __restrict__ mem, const float* __restrict__ nf,
           float* __restrict__ raw, float* __restrict__ den) {
  int idx = blockIdx.x * 256 + threadIdx.x;
  if (idx >= SL_ * E_) return;
  int s = idx / E_;
  int a = st[idx];
  int b = en[idx];
  float w = expf((ts[idx] - bt[s]) * (1.0f / 3600.0f));
  float* rs = raw + ((size_t)a * SL_ + s) * RAW_;         // msg_s = [mem[b], nf[b]] into seg of starts
  float* re = raw + ((size_t)b * SL_ + s) * RAW_ + 192;   // msg_e = [mem[a], nf[a]] into seg of ends
  const float* mb = mem + (size_t)b * EMB_;
  const float* ma = mem + (size_t)a * EMB_;
  const float* fb = nf + (size_t)b * F_;
  const float* fa = nf + (size_t)a * F_;
  for (int c = 0; c < EMB_; ++c) {
    atomicAdd(rs + c, w * mb[c]);
    atomicAdd(re + c, w * ma[c]);
  }
  for (int c = 0; c < F_; ++c) {
    atomicAdd(rs + EMB_ + c, w * fb[c]);
    atomicAdd(re + EMB_ + c, w * fa[c]);
  }
  atomicAdd(den + (size_t)s * N_ + a, w);
  atomicAdd(den + (size_t)(SL_ + s) * N_ + b, w);
}

__global__ void __launch_bounds__(256)
normalize_raw(float* __restrict__ raw, const float* __restrict__ den) {
  long idx = (long)blockIdx.x * 256 + threadIdx.x;
  if (idx >= (long)N_ * SL_ * RAW_) return;
  int c = (int)(idx % RAW_);
  long ns = idx / RAW_;
  int s = (int)(ns % SL_);
  int n = (int)(ns / SL_);
  float d = (c < 192) ? den[(size_t)s * N_ + n] : den[(size_t)(SL_ + s) * N_ + n];
  float inv = (d > 0.0f) ? 1.0f / fmaxf(d, 1e-30f) : 0.0f;
  raw[idx] *= inv;
}

__global__ void __launch_bounds__(256)
appear_kernel(const float* __restrict__ den, float* __restrict__ ap) {
  int n = blockIdx.x * 256 + threadIdx.x;
  if (n >= N_) return;
  float a = 0.0f;
  for (int s = 0; s < SL_; ++s) {
    if (den[(size_t)s * N_ + n] > 0.0f || den[(size_t)(SL_ + s) * N_ + n] > 0.0f) a = 1.0f;
  }
  ap[n] = a;
}

// ---------------- bf16 WMMA GEMM: C[M,Nc] = act(A_f32[M,K] @ W[K,Nc] + bias) ----------------
// grid = (M/16, Nc/128); block = 256 (8 waves); wave w owns 16x16 col-tile (by*8+w).
// A tile: one float4 global load per thread, packed to bf16 and scattered into the WMMA
// A-operand lane layout in LDS (4 consecutive aligned k-cols map to 4 consecutive LDS
// slots -> one ds_store_b64).  Each lane then reads its operand with 2x ds_load_b128.
// B comes pre-swizzled (repack_w): 2x global_load_b128 per lane per K-step.
// K unrolled by 2: one LDS fill + barrier pair feeds two v_wmma issues.
__global__ void __launch_bounds__(256)
wmma_gemm(const float* __restrict__ A, const __bf16* __restrict__ Bsw,
          const float* __restrict__ bias, float* __restrict__ C,
          int M, int K, int Nc, int act) {
  __shared__ __align__(32) __bf16 Asw[32][32];   // [lane][2 k-blocks x 16 elems]
  const int wave  = threadIdx.x >> 5;
  const int lane  = threadIdx.x & 31;
  const int l16   = lane & 15;
  const int khalf = lane >> 4;
  const int rowBase = blockIdx.x * 16;
  const int ct      = blockIdx.y * 8 + wave;
  const int nkb     = K >> 5;

  // per-thread A-fill coordinates (one float4 per thread covers the 16x64 tile)
  const int fr  = threadIdx.x >> 4;         // row 0..15
  const int fc  = (threadIdx.x & 15) * 4;   // k-col 0..60, step 4
  const int fkb = fc >> 5;
  const int fcc = fc & 31;
  const int fkh = fcc >> 4;
  const int fe0 = fcc & 15;
  unsigned* fdst = (unsigned*)&Asw[fr + 16 * fkh][fkb * 16 + fe0];

  v8f acc = {};
  for (int k0 = 0; k0 < K; k0 += 64) {
    // B operands: pre-swizzled, contiguous 32B per lane per k-block
    const int kb = k0 >> 5;
    v16bf b0 = *(const v16bf*)(Bsw + (((size_t)ct * nkb + kb) * 32 + lane) * 16);
    v16bf b1 = *(const v16bf*)(Bsw + (((size_t)ct * nkb + kb + 1) * 32 + lane) * 16);
    // cooperative A tile: 16 rows x 64 k, one float4 -> one ds_store_b64 per thread
    const float4 av = *(const float4*)(A + (size_t)(rowBase + fr) * K + k0 + fc);
    fdst[0] = pack2(av.x, av.y);
    fdst[1] = pack2(av.z, av.w);
    __syncthreads();
    v16bf a0 = *(const v16bf*)&Asw[lane][0];
    v16bf a1 = *(const v16bf*)&Asw[lane][16];
    acc = __builtin_amdgcn_wmma_f32_16x16x32_bf16(false, a0, false, b0,
                                                  (short)0, acc, false, false);
    acc = __builtin_amdgcn_wmma_f32_16x16x32_bf16(false, a1, false, b1,
                                                  (short)0, acc, false, false);
    __syncthreads();
  }
  const int col = ct * 16 + l16;
  const float bs = bias ? bias[col] : 0.0f;
  #pragma unroll
  for (int v = 0; v < 8; ++v) {
    int row = rowBase + khalf * 8 + v;    // D layout: lane half selects M block
    float val = acc[v] + bs;
    if (act == 1) val = gelu_exact(val);
    C[(size_t)row * Nc + col] = val;
  }
}

// ---------------- LayerNorm over last dim 128, one wave per row (optional residual add) ----------------
__global__ void __launch_bounds__(256)
ln_rows(const float* __restrict__ A, const float* __restrict__ Add,
        float* __restrict__ Out, int R) {
  int gid  = (blockIdx.x * blockDim.x + threadIdx.x) >> 5;
  int lane = threadIdx.x & 31;
  if (gid >= R) return;
  float4 x = *(const float4*)(A + (size_t)gid * 128 + lane * 4);
  if (Add) {
    float4 y = *(const float4*)(Add + (size_t)gid * 128 + lane * 4);
    x.x += y.x; x.y += y.y; x.z += y.z; x.w += y.w;
  }
  float s  = wave_sum(x.x + x.y + x.z + x.w);
  float sq = wave_sum(x.x * x.x + x.y * x.y + x.z * x.z + x.w * x.w);
  float mu  = s * (1.0f / 128.0f);
  float var = sq * (1.0f / 128.0f) - mu * mu;
  float inv = rsqrtf(var + 1e-5f);
  float4 o;
  o.x = (x.x - mu) * inv; o.y = (x.y - mu) * inv;
  o.z = (x.z - mu) * inv; o.w = (x.w - mu) * inv;
  *(float4*)(Out + (size_t)gid * 128 + lane * 4) = o;
}

// ---------------- time-mixing MLP along SL (K=24, pure VALU; weights in LDS) ----------------
__global__ void __launch_bounds__(256)
time_mix(const float* __restrict__ xn, const float* __restrict__ x,
         const float* __restrict__ W1, const float* __restrict__ B1,
         const float* __restrict__ W2, const float* __restrict__ B2,
         float* __restrict__ at) {
  __shared__ float sW1[SL_ * SL_], sW2[SL_ * SL_], sB1[SL_], sB2[SL_];
  for (int i = threadIdx.x; i < SL_ * SL_; i += 256) { sW1[i] = W1[i]; sW2[i] = W2[i]; }
  if (threadIdx.x < SL_) { sB1[threadIdx.x] = B1[threadIdx.x]; sB2[threadIdx.x] = B2[threadIdx.x]; }
  __syncthreads();
  int idx = blockIdx.x * 256 + threadIdx.x;
  if (idx >= N_ * EMB_) return;
  int n = idx >> 7, c = idx & 127;
  float xs[SL_];
  #pragma unroll
  for (int s = 0; s < SL_; ++s) xs[s] = xn[((size_t)n * SL_ + s) * EMB_ + c];
  float g[SL_];
  #pragma unroll
  for (int s2 = 0; s2 < SL_; ++s2) {
    float acc = sB1[s2];
    #pragma unroll
    for (int s = 0; s < SL_; ++s) acc += xs[s] * sW1[s * SL_ + s2];
    g[s2] = gelu_exact(acc);
  }
  #pragma unroll
  for (int s2 = 0; s2 < SL_; ++s2) {
    float acc = sB2[s2];
    #pragma unroll
    for (int s = 0; s < SL_; ++s) acc += g[s] * sW2[s * SL_ + s2];
    size_t o = ((size_t)n * SL_ + s2) * EMB_ + c;
    at[o] = acc + x[o];
  }
}

__global__ void __launch_bounds__(256)
daymean_kernel(const float* __restrict__ ac, float* __restrict__ dm) {
  int idx = blockIdx.x * 256 + threadIdx.x;
  if (idx >= N_ * EMB_) return;
  int n = idx >> 7, c = idx & 127;
  float s = 0.0f;
  for (int t = 0; t < SL_; ++t) s += ac[((size_t)n * SL_ + t) * EMB_ + c];
  dm[idx] = s * (1.0f / (float)SL_);
}

// ---------------- GRU combine + appeared select ----------------
__global__ void __launch_bounds__(256)
gru_kernel(const float* __restrict__ gi, const float* __restrict__ gh,
           const float* __restrict__ mem, const float* __restrict__ ap,
           float* __restrict__ ne) {
  int idx = blockIdx.x * 256 + threadIdx.x;
  if (idx >= N_ * EMB_) return;
  int n = idx >> 7, c = idx & 127;
  size_t base = (size_t)n * 384;
  float r  = 1.0f / (1.0f + expf(-(gi[base + c] + gh[base + c])));
  float z  = 1.0f / (1.0f + expf(-(gi[base + 128 + c] + gh[base + 128 + c])));
  float nn = tanhf(gi[base + 256 + c] + r * gh[base + 256 + c]);
  float m  = mem[idx];
  float h  = (1.0f - z) * nn + z * m;
  ne[idx] = (ap[n] > 0.5f) ? h : m;
}

// ---------------- label statistics (both tasks at once) ----------------
__global__ void __launch_bounds__(256)
stats_kernel(const float* __restrict__ lab, float* __restrict__ st) {
  int idx = blockIdx.x * 256 + threadIdx.x;
  float s0 = 0.f, q0 = 0.f, s1 = 0.f, q1 = 0.f;
  if (idx < SL_ * N_) {
    float a = lab[(size_t)idx * NT_ + 0];
    float b = lab[(size_t)idx * NT_ + 1];
    s0 = a; q0 = a * a; s1 = b; q1 = b * b;
  }
  s0 = wave_sum(s0); q0 = wave_sum(q0); s1 = wave_sum(s1); q1 = wave_sum(q1);
  if ((threadIdx.x & 31) == 0) {
    atomicAdd(st + 0, s0); atomicAdd(st + 1, q0);
    atomicAdd(st + 4, s1); atomicAdd(st + 5, q1);
  }
}

__global__ void finalize_stats(float* st) {
  const float invc = 1.0f / (float)(SL_ * N_);
  for (int t = 0; t < NT_; ++t) {
    float mu  = st[t * 4 + 0] * invc;
    float var = st[t * 4 + 1] * invc - mu * mu;
    st[t * 4 + 2] = mu;
    st[t * 4 + 3] = sqrtf(fmaxf(var, 0.0f));
  }
}

// ---------------- precompute wv1 = W_v @ W_o1, bv1 = b_v @ W_o1 ----------------
__global__ void prep_mlp(const float* __restrict__ Wv, const float* __restrict__ bv,
                         const float* __restrict__ Wo1, float* __restrict__ out) {
  int j = threadIdx.x;
  if (j >= H_) return;
  float a = 0.f, b = 0.f;
  for (int d = 0; d < EMB_; ++d) {
    float w = Wo1[d * H_ + j];
    a += Wv[d] * w;
    b += bv[d] * w;
  }
  out[j] = a;
  out[H_ + j] = b;
}

// ---------------- attention + output head: one wave per (node, task) ----------------
__global__ void __launch_bounds__(256)
attention_kernel(const float* __restrict__ Q, const float* __restrict__ Kb,
                 const float* __restrict__ lab, const float* __restrict__ Wk2,
                 const float* __restrict__ bk2, const float* __restrict__ st,
                 const float* __restrict__ wvb, const float* __restrict__ bo1,
                 const float* __restrict__ Wo2, const float* __restrict__ bo2,
                 const float* __restrict__ Wo3, const float* __restrict__ bo3,
                 float* __restrict__ res, float* __restrict__ att) {
  int gid  = (blockIdx.x * blockDim.x + threadIdx.x) >> 5;
  int lane = threadIdx.x & 31;
  if (gid >= N_ * NT_) return;
  int n = gid >> 1, task = gid & 1;
  float mu = st[task * 4 + 2], sd = st[task * 4 + 3];
  float invsd = 1.0f / sd;
  const float4 q  = *(const float4*)(Q + (size_t)n * 128 + lane * 4);
  const float4 w2 = *(const float4*)(Wk2 + lane * 4);
  const float4 b2 = *(const float4*)(bk2 + lane * 4);
  float qw = wave_sum(q.x * w2.x + q.y * w2.y + q.z * w2.z + q.w * w2.w);
  float qb = wave_sum(q.x * b2.x + q.y * b2.y + q.z * b2.z + q.w * b2.w);
  const float scale = 0.08838834764831845f;  // 1/sqrt(128)
  float sc[SL_];
  #pragma unroll
  for (int t = 0; t < SL_; ++t) {
    const float4 k = *(const float4*)(Kb + ((size_t)t * N_ + n) * 128 + lane * 4);
    float s0 = wave_sum(q.x * k.x + q.y * k.y + q.z * k.z + q.w * k.w);
    float l  = lab[((size_t)t * N_ + n) * NT_ + task];
    sc[t] = (s0 + (l - mu) * invsd * qw + qb) * scale;
  }
  float m = -1e30f;
  #pragma unroll
  for (int t = 0; t < SL_; ++t) m = fmaxf(m, sc[t]);
  float Z = 0.f;
  #pragma unroll
  for (int t = 0; t < SL_; ++t) { sc[t] = expf(sc[t] - m); Z += sc[t]; }
  float invZ = 1.0f / Z;
  float a_lab = 0.f, attv = 0.f;
  #pragma unroll
  for (int t = 0; t < SL_; ++t) {
    float e = sc[t] * invZ;
    float l = lab[((size_t)t * N_ + n) * NT_ + task];
    a_lab += e * (l - mu) * invsd;
    if (lane == t) attv = e;
  }
  if (lane < SL_) att[(size_t)(task * SL_ + lane) * N_ + n] = attv;
  // out = a_lab*W_v + b_v  ->  o1 = relu(a_lab*wv1 + bv1 + b_o1)
  float acc_a = 0.f, acc_b = 0.f;
  for (int k = 0; k < H_; ++k) {
    float o1 = fmaxf(a_lab * wvb[k] + wvb[H_ + k] + bo1[k], 0.0f);
    acc_a += o1 * Wo2[k * H_ + lane];
    acc_b += o1 * Wo2[k * H_ + lane + 32];
  }
  float o2a = fmaxf(acc_a + bo2[lane], 0.0f);
  float o2b = fmaxf(acc_b + bo2[lane + 32], 0.0f);
  float part = wave_sum(o2a * Wo3[lane] + o2b * Wo3[lane + 32]);
  if (lane == 0) res[(size_t)n * NT_ + task] = (part + bo3[0]) * sd + mu;
}

// ---------------- driver ----------------
extern "C" void kernel_launch(void* const* d_in, const int* in_sizes, int n_in,
                              void* d_out, int out_size, void* d_ws, size_t ws_size,
                              hipStream_t stream) {
  (void)in_sizes; (void)n_in; (void)out_size; (void)ws_size;
  float* fw = (float*)d_ws;

  const int*   t_st   = (const int*)d_in[0];
  const int*   t_en   = (const int*)d_in[1];
  const float* t_ts   = (const float*)d_in[2];
  const float* bt     = (const float*)d_in[3];
  const float* mem    = (const float*)d_in[4];
  const float* nf     = (const float*)d_in[5];
  const float* ctx    = (const float*)d_in[6];
  const float* labels = (const float*)d_in[7];
  const float* W_cin = (const float*)d_in[8];
  const float* b_cin = (const float*)d_in[9];
  const float* W_t1  = (const float*)d_in[10];
  const float* b_t1  = (const float*)d_in[11];
  const float* W_t2  = (const float*)d_in[12];
  const float* b_t2  = (const float*)d_in[13];
  const float* W_c1  = (const float*)d_in[14];
  const float* b_c1  = (const float*)d_in[15];
  const float* W_c2  = (const float*)d_in[16];
  const float* b_c2  = (const float*)d_in[17];
  const float* W_ih  = (const float*)d_in[18];
  const float* b_ih  = (const float*)d_in[19];
  const float* W_hh  = (const float*)d_in[20];
  const float* b_hh  = (const float*)d_in[21];
  const float* W_q   = (const float*)d_in[22];
  const float* b_q   = (const float*)d_in[23];
  const float* W_k   = (const float*)d_in[24];
  const float* b_k   = (const float*)d_in[25];
  const float* W_k2  = (const float*)d_in[26];
  const float* b_k2  = (const float*)d_in[27];
  const float* W_v   = (const float*)d_in[28];
  const float* b_v   = (const float*)d_in[29];
  const float* W_o1  = (const float*)d_in[30];
  const float* b_o1  = (const float*)d_in[31];
  const float* W_o2  = (const float*)d_in[32];
  const float* b_o2  = (const float*)d_in[33];
  const float* W_o3  = (const float*)d_in[34];
  const float* b_o3  = (const float*)d_in[35];

  float* out     = (float*)d_out;
  float* out_res = out;                        // [N,2]
  float* out_ne  = out + (size_t)N_ * NT_;     // [N,128]
  float* out_att = out_ne + (size_t)N_ * EMB_; // [2*SL, N]

  __bf16* wb     = (__bf16*)(fw + WB_OFF);
  __bf16* Wcin_b = wb;
  __bf16* Wc1_b  = wb + 49152;
  __bf16* Wc2_b  = wb + 65536;
  __bf16* Wih_b  = wb + 81920;
  __bf16* Whh_b  = wb + 131072;
  __bf16* Wq_b   = wb + 180224;
  __bf16* Wk_b   = wb + 196608;

  // deterministic re-zero of all atomic accumulators (graph-capture legal)
  hipMemsetAsync(fw + RAW_OFF, 0, (size_t)N_ * SL_ * RAW_ * sizeof(float), stream);
  hipMemsetAsync(fw + DEN_OFF, 0, (size_t)(2 * SL_ * N_ + N_) * sizeof(float), stream);
  hipMemsetAsync(fw + STAT_OFF, 0, 16 * sizeof(float), stream);

  // repack weights into the WMMA B-operand lane layout (bf16)
  repack_w<<<192, 256, 0, stream>>>(W_cin, Wcin_b, RAW_, EMB_);
  repack_w<<<64, 256, 0, stream>>>(W_c1, Wc1_b, EMB_, EMB_);
  repack_w<<<64, 256, 0, stream>>>(W_c2, Wc2_b, EMB_, EMB_);
  repack_w<<<192, 256, 0, stream>>>(W_ih, Wih_b, EMB_, 3 * EMB_);
  repack_w<<<192, 256, 0, stream>>>(W_hh, Whh_b, EMB_, 3 * EMB_);
  repack_w<<<64, 256, 0, stream>>>(W_q, Wq_b, EMB_, EMB_);
  repack_w<<<64, 256, 0, stream>>>(W_k, Wk_b, EMB_, EMB_);

  agg_kernel<<<(SL_ * E_ + 255) / 256, 256, 0, stream>>>(
      t_st, t_en, t_ts, bt, mem, nf, fw + RAW_OFF, fw + DEN_OFF);
  normalize_raw<<<(int)(((long)N_ * SL_ * RAW_ + 255) / 256), 256, 0, stream>>>(
      fw + RAW_OFF, fw + DEN_OFF);
  appear_kernel<<<(N_ + 255) / 256, 256, 0, stream>>>(fw + DEN_OFF, fw + APPEAR_OFF);

  const int R = N_ * SL_;  // 480000 rows
  // x = raw @ W_cin + b_cin
  wmma_gemm<<<dim3(R / 16, 1), 256, 0, stream>>>(fw + RAW_OFF, Wcin_b, b_cin, fw + X_OFF, R, RAW_, EMB_, 0);
  ln_rows<<<R / 8, 256, 0, stream>>>(fw + X_OFF, nullptr, fw + A_OFF, R);
  time_mix<<<(N_ * EMB_) / 256, 256, 0, stream>>>(fw + A_OFF, fw + X_OFF, W_t1, b_t1, W_t2, b_t2, fw + AT_OFF);
  ln_rows<<<R / 8, 256, 0, stream>>>(fw + AT_OFF, nullptr, fw + A_OFF, R);
  // h = gelu(ln(at) @ W_c1 + b_c1) @ W_c2 + b_c2
  wmma_gemm<<<dim3(R / 16, 1), 256, 0, stream>>>(fw + A_OFF, Wc1_b, b_c1, fw + B_OFF, R, EMB_, EMB_, 1);
  wmma_gemm<<<dim3(R / 16, 1), 256, 0, stream>>>(fw + B_OFF, Wc2_b, b_c2, fw + A_OFF, R, EMB_, EMB_, 0);
  // across_channel = ln(h + across_time); day_message = mean over SL
  ln_rows<<<R / 8, 256, 0, stream>>>(fw + A_OFF, fw + AT_OFF, fw + B_OFF, R);
  daymean_kernel<<<(N_ * EMB_) / 256, 256, 0, stream>>>(fw + B_OFF, fw + DM_OFF);
  // GRU
  wmma_gemm<<<dim3(N_ / 16, 3), 256, 0, stream>>>(fw + DM_OFF, Wih_b, b_ih, fw + GI_OFF, N_, EMB_, 3 * EMB_, 0);
  wmma_gemm<<<dim3(N_ / 16, 3), 256, 0, stream>>>(mem, Whh_b, b_hh, fw + GH_OFF, N_, EMB_, 3 * EMB_, 0);
  gru_kernel<<<(N_ * EMB_) / 256, 256, 0, stream>>>(fw + GI_OFF, fw + GH_OFF, mem, fw + APPEAR_OFF, out_ne);
  // Q and Kbase
  wmma_gemm<<<dim3(N_ / 16, 1), 256, 0, stream>>>(out_ne, Wq_b, b_q, fw + Q_OFF, N_, EMB_, EMB_, 0);
  wmma_gemm<<<dim3(R / 16, 1), 256, 0, stream>>>(ctx, Wk_b, b_k, fw + KB_OFF, R, EMB_, EMB_, 0);
  // label stats + attention head
  stats_kernel<<<(SL_ * N_ + 255) / 256, 256, 0, stream>>>(labels, fw + STAT_OFF);
  finalize_stats<<<1, 1, 0, stream>>>(fw + STAT_OFF);
  prep_mlp<<<1, 64, 0, stream>>>(W_v, b_v, W_o1, fw + WV1_OFF);
  attention_kernel<<<(N_ * NT_) / 8, 256, 0, stream>>>(
      fw + Q_OFF, fw + KB_OFF, labels, W_k2, b_k2, fw + STAT_OFF, fw + WV1_OFF,
      b_o1, W_o2, b_o2, W_o3, b_o3, out_res, out_att);
}
